// LSTMDecoder_25185688224042
// MI455X (gfx1250) — matrix operand order repaired and checked by reading, not compiled, not run on hardware
//
#include <hip/hip_runtime.h>
#include <hip/hip_bf16.h>

typedef __attribute__((ext_vector_type(16))) __bf16 v16bf;
typedef __attribute__((ext_vector_type(8)))  __bf16 v8bf;
typedef __attribute__((ext_vector_type(8)))  float  v8f;

#define BATCH 4096
#define NNODE 20
#define DIM   256
#define MSTEPS 30
#define TOURW 31
#define RWG   64      // batch rows per workgroup

static __device__ __forceinline__ unsigned short f2bf(float f){
  unsigned u = __float_as_uint(f);
  unsigned r = (u + 0x7FFFu + ((u >> 16) & 1u)) >> 16;   // RNE
  return (unsigned short)r;
}
static __device__ __forceinline__ float bf2f_lo(unsigned packed){   // bf16 in low half
  return __uint_as_float(packed << 16);
}
static __device__ __forceinline__ float bf2f_hi(unsigned packed){   // bf16 in high half
  return __uint_as_float(packed & 0xFFFF0000u);
}
static __device__ __forceinline__ __bf16 f2bfe(float f){
  unsigned short u = f2bf(f); __bf16 r; __builtin_memcpy(&r, &u, 2); return r;
}
static __device__ __forceinline__ float sigm(float x){ return 1.f/(1.f+__expf(-x)); }

static __device__ __forceinline__ v8f wmma_bf16(v16bf a, v16bf b, v8f c){
  // D = A(16x32 bf16) * B(32x16 bf16) + C(16x16 f32)
  return __builtin_amdgcn_wmma_f32_16x16x32_bf16(false, a, false, b, (short)0, c, false, false);
}
static __device__ __forceinline__ v16bf cat8(v8bf lo, v8bf hi){
  return __builtin_shufflevector(lo, hi, 0,1,2,3,4,5,6,7,8,9,10,11,12,13,14,15);
}

// ---------------------------------------------------------------------------
// Kernel 1: pack W = [W_ih | W_hh] (1024 x 512) and attn_w (256 x 256) into
// the CDNA5 wave32 WMMA B-matrix lane layout (bf16), tiles of 32(K) x 16(N).
// B layout: half j, lane L -> K = ktile*32 + (j&7) + (j>=8?16:0) + (L<16?0:8), N = ntile*16 + L%16
// ---------------------------------------------------------------------------
__global__ void pack_weights_kernel(const float* __restrict__ W_ih,
                                    const float* __restrict__ W_hh,
                                    const float* __restrict__ attn_w,
                                    unsigned short* __restrict__ wpack,
                                    unsigned short* __restrict__ apack){
  int e = blockIdx.x * 256 + threadIdx.x;
  const int NW = 16 * 64 * 32 * 16;      // 512K elems: K=512 (16 ktiles) x N=1024 (64 ntiles)
  if(e < NW){
    int j = e & 15, lane = (e >> 4) & 31, nt = (e >> 9) & 63, kt = e >> 15;
    int K = kt * 32 + (j & 7) + ((j >> 3) << 4) + ((lane < 16) ? 0 : 8);
    int G = nt * 16 + (lane & 15);
    float v = (K < DIM) ? W_ih[G * DIM + K] : W_hh[G * DIM + (K - DIM)];
    wpack[e] = f2bf(v);
  }
  int e2 = e - NW;
  if(e2 >= 0 && e2 < 8 * 16 * 32 * 16){  // 64K elems: K=256 x N=256
    int j = e2 & 15, lane = (e2 >> 4) & 31, nt = (e2 >> 9) & 15, kt = e2 >> 13;
    int K = kt * 32 + (j & 7) + ((j >> 3) << 4) + ((lane < 16) ? 0 : 8);
    int d = nt * 16 + (lane & 15);
    apack[e2] = f2bf(attn_w[d * DIM + K]);
  }
}

// ---------------------------------------------------------------------------
// Kernel 2: keys = node_emb @ attn_w^T + attn_b  as [B*N, 256] f32.
// One wave per 16-row M-tile (5120 tiles). 2-stage software pipeline:
// B tiles (and next-kt A tile) are loaded BEFORE the WMMAs that consume the
// previous set, so global/ds latency overlaps the matrix pipe.
// ---------------------------------------------------------------------------
__global__ void __launch_bounds__(256) keys_kernel(const float* __restrict__ node_emb,
                                                   const unsigned short* __restrict__ apack_u,
                                                   const float* __restrict__ attn_b,
                                                   float* __restrict__ keys){
  const int lane = threadIdx.x & 31;
  const int mt   = blockIdx.x * 8 + (threadIdx.x >> 5);   // 640 blocks * 8 waves = 5120
  const v16bf* ap = reinterpret_cast<const v16bf*>(apack_u);
  const int rowA = mt * 16 + (lane & 15);

  v8f acc[16];
  #pragma unroll
  for(int n = 0; n < 16; n++){
    #pragma unroll
    for(int r = 0; r < 8; r++) acc[n][r] = 0.f;
  }

  // A-tile loader: lane L row M=L%16; halves 0..7 -> K=c0..+7, 8..15 -> K=c0+16..+23
  auto loadA = [&](int kt) -> v16bf {
    const float* src = node_emb + rowA * DIM + kt * 32 + ((lane < 16) ? 0 : 8);
    v16bf a;
    #pragma unroll
    for(int i = 0; i < 8; i++){ a[i] = f2bfe(src[i]); a[i + 8] = f2bfe(src[16 + i]); }
    return a;
  };

  // prologue: A(0) and B(kt=0, ng=0)
  v16bf a = loadA(0);
  v16bf an = a;
  const v16bf* bp0 = ap + lane;
  v16bf nb0 = bp0[0 * 32], nb1 = bp0[1 * 32], nb2 = bp0[2 * 32], nb3 = bp0[3 * 32];

  #pragma unroll 1
  for(int kt = 0; kt < 8; kt++){
    #pragma unroll
    for(int ng = 0; ng < 4; ng++){
      v16bf b0 = nb0, b1 = nb1, b2 = nb2, b3 = nb3;
      // prefetch next group's tiles before consuming current ones
      if(ng < 3){
        const v16bf* bp = ap + (kt * 16 + (ng + 1) * 4) * 32 + lane;
        nb0 = bp[0]; nb1 = bp[32]; nb2 = bp[64]; nb3 = bp[96];
      } else if(kt < 7){
        const v16bf* bp = ap + ((kt + 1) * 16) * 32 + lane;
        nb0 = bp[0]; nb1 = bp[32]; nb2 = bp[64]; nb3 = bp[96];
        an = loadA(kt + 1);
      }
      acc[ng * 4 + 0] = wmma_bf16(a, b0, acc[ng * 4 + 0]);
      acc[ng * 4 + 1] = wmma_bf16(a, b1, acc[ng * 4 + 1]);
      acc[ng * 4 + 2] = wmma_bf16(a, b2, acc[ng * 4 + 2]);
      acc[ng * 4 + 3] = wmma_bf16(a, b3, acc[ng * 4 + 3]);
    }
    a = an;
  }

  #pragma unroll
  for(int nt = 0; nt < 16; nt++){
    int d = nt * 16 + (lane & 15);
    float bias = attn_b[d];
    #pragma unroll
    for(int r = 0; r < 8; r++){
      int row = mt * 16 + r + ((lane < 16) ? 0 : 8);    // C/D layout: M = r (+8 for hi lanes)
      keys[row * DIM + d] = acc[nt][r] + bias;
    }
  }
}

// ---------------------------------------------------------------------------
// Kernel 3: persistent pointer-LSTM decoder. 64 batch rows / workgroup.
// 8 waves: wave w -> M-tile mt=w/2 (16 rows), d-tile group (w&1)*8..+7.
// gates = [x|h](bf16, LDS) @ Wpack with 2-stage pipelined loads; fused LSTM
// elementwise per d-tile; c kept in VGPRs; h double-buffered in LDS.
// ---------------------------------------------------------------------------
__global__ void __launch_bounds__(256) decoder_kernel(
    const float* __restrict__ node_emb,
    const unsigned short* __restrict__ wpack_u,
    const float* __restrict__ b_ih, const float* __restrict__ b_hh,
    const float* __restrict__ done_w, const float* __restrict__ done_b,
    const float* __restrict__ revisit_p,
    const float* __restrict__ keys,
    float* __restrict__ out)
{
  __shared__ unsigned short Xx[RWG * DIM];        // 32 KB  (x part, bf16)
  __shared__ unsigned short Xh[2][RWG * DIM];     // 64 KB  (h part, double buffered)
  __shared__ float doneW[DIM];                    // staged done_w
  __shared__ float scoreL[RWG][NNODE];
  __shared__ float doneL[RWG];
  __shared__ unsigned visA[RWG];
  __shared__ int uniqA[RWG], firstA[RWG], currA[RWG], compA[RWG];

  const int tid  = threadIdx.x;
  const int lane = tid & 31;
  const int w    = tid >> 5;
  const int bbase = blockIdx.x * RWG;
  const int mt   = w >> 1;
  const int ntb  = (w & 1) * 8;
  const v16bf* wp = reinterpret_cast<const v16bf*>(wpack_u);
  const float rp = revisit_p[0];
  const float db = done_b[0];
  float* tours = out;
  float* logps = out + BATCH * TOURW;

  // warm the packed weights into cache (streamed 30x by this workgroup)
  #pragma unroll 1
  for(int i = tid; i < 4096; i += 256)
    __builtin_prefetch(wpack_u + i * 128, 0, 0);   // 256B granules over 1 MB

  // init: x = mean over nodes, h = 0, state = 0, stage done_w
  for(int i = tid; i < RWG * DIM; i += 256){
    int r = i >> 8, c = i & 255;
    const float* p = node_emb + ((bbase + r) * NNODE) * DIM + c;
    float s = 0.f;
    #pragma unroll 1
    for(int n = 0; n < NNODE; n++) s += p[n * DIM];
    Xx[i] = f2bf(s * (1.f / NNODE));
    Xh[0][i] = 0; Xh[1][i] = 0;
  }
  doneW[tid] = done_w[tid];
  if(tid < RWG){ visA[tid] = 0; uniqA[tid] = 0; firstA[tid] = 0; currA[tid] = 0; compA[tid] = 0; }

  // hoisted gate biases: per d-tile this lane owns column nt*16 + lane%16
  float biasI[8], biasF[8], biasG[8], biasO[8];
  #pragma unroll
  for(int lnt = 0; lnt < 8; lnt++){
    int col = (ntb + lnt) * 16 + (lane & 15);
    biasI[lnt] = b_ih[col]           + b_hh[col];
    biasF[lnt] = b_ih[DIM + col]     + b_hh[DIM + col];
    biasG[lnt] = b_ih[2 * DIM + col] + b_hh[2 * DIM + col];
    biasO[lnt] = b_ih[3 * DIM + col] + b_hh[3 * DIM + col];
  }

  v8f cst[8];                       // c-state: 8 d-tiles x 8 rows per lane
  #pragma unroll
  for(int i = 0; i < 8; i++){
    #pragma unroll
    for(int r = 0; r < 8; r++) cst[i][r] = 0.f;
  }

  const int rowA = mt * 16 + (lane & 15);
  const int offA = (lane < 16) ? 0 : 8;

  int cur = 0;
  for(int t = 0; t < MSTEPS; t++){
    __syncthreads();
    const unsigned short* Xhc = Xh[cur];
    unsigned short* Xhn = Xh[cur ^ 1];

    // A-tile loader from LDS [x|h] (two ds_load_b128 per lane)
    auto loadA = [&](int kt) -> v16bf {
      int off = kt * 32 + offA;
      const unsigned short* src = (kt < 8) ? (Xx  + rowA * DIM + off)
                                           : (Xhc + rowA * DIM + (off - DIM));
      v8bf lo = *reinterpret_cast<const v8bf*>(src);
      v8bf hi = *reinterpret_cast<const v8bf*>(src + 16);
      return cat8(lo, hi);
    };

    // ---- gates GEMM (WMMA, 2-stage pipelined) + fused LSTM elementwise ----
    #pragma unroll
    for(int lnt = 0; lnt < 8; lnt++){
      const int nt = ntb + lnt;
      v8f ai, af, ag, ao;
      #pragma unroll
      for(int r = 0; r < 8; r++){ ai[r]=0.f; af[r]=0.f; ag[r]=0.f; ao[r]=0.f; }

      // prologue for kt=0
      v16bf a = loadA(0);
      const v16bf* wb0 = wp + nt * 32 + lane;
      v16bf nb0 = wb0[0], nb1 = wb0[16*32], nb2 = wb0[32*32], nb3 = wb0[48*32];

      #pragma unroll 1
      for(int kt = 0; kt < 16; kt++){
        v16bf ac = a;
        v16bf b0 = nb0, b1 = nb1, b2 = nb2, b3 = nb3;
        if(kt < 15){
          a = loadA(kt + 1);
          const v16bf* wb = wp + (kt + 1) * 64 * 32 + nt * 32 + lane;
          nb0 = wb[0]; nb1 = wb[16*32]; nb2 = wb[32*32]; nb3 = wb[48*32];
        }
        ai = wmma_bf16(ac, b0, ai);
        af = wmma_bf16(ac, b1, af);
        ag = wmma_bf16(ac, b2, ag);
        ao = wmma_bf16(ac, b3, ao);
      }
      const int col = nt * 16 + (lane & 15);
      const int rb = mt * 16 + ((lane < 16) ? 0 : 8);
      #pragma unroll
      for(int r = 0; r < 8; r++){
        float iv = sigm(ai[r] + biasI[lnt]);
        float fv = sigm(af[r] + biasF[lnt]);
        float gv = tanhf(ag[r] + biasG[lnt]);
        float ov = sigm(ao[r] + biasO[lnt]);
        float c2 = fv * cst[lnt][r] + iv * gv;
        cst[lnt][r] = c2;
        float h2 = ov * tanhf(c2);
        Xhn[(rb + r) * DIM + col] = f2bf(h2);
      }
    }
    __syncthreads();

    // ---- pointer scores: 4 threads/row, 5 nodes each; q==0 also done-logit ----
    {
      const int q = tid & 3, r = tid >> 2;
      const int b = bbase + r;
      const float4* k0 = (const float4*)(keys + (b * NNODE + q * 5) * DIM);
      const float4* k1 = k0 + DIM / 4;
      const float4* k2 = k0 + 2 * (DIM / 4);
      const float4* k3 = k0 + 3 * (DIM / 4);
      const float4* k4 = k0 + 4 * (DIM / 4);
      const uint2* hrow = (const uint2*)(Xhn + r * DIM);       // 4 bf16 per uint2
      const float4* dw4 = (const float4*)doneW;
      float s0=0.f, s1=0.f, s2=0.f, s3=0.f, s4=0.f, dacc=0.f;
      #pragma unroll 1
      for(int kk = 0; kk < DIM / 4; kk++){
        uint2 hb = hrow[kk];
        float h0 = bf2f_lo(hb.x), h1 = bf2f_hi(hb.x);
        float h2v = bf2f_lo(hb.y), h3 = bf2f_hi(hb.y);
        float4 a0 = k0[kk]; s0 += h0*a0.x + h1*a0.y + h2v*a0.z + h3*a0.w;
        float4 a1 = k1[kk]; s1 += h0*a1.x + h1*a1.y + h2v*a1.z + h3*a1.w;
        float4 a2 = k2[kk]; s2 += h0*a2.x + h1*a2.y + h2v*a2.z + h3*a2.w;
        float4 a3 = k3[kk]; s3 += h0*a3.x + h1*a3.y + h2v*a3.z + h3*a3.w;
        float4 a4 = k4[kk]; s4 += h0*a4.x + h1*a4.y + h2v*a4.z + h3*a4.w;
        if(q == 0){ float4 d4 = dw4[kk]; dacc += h0*d4.x + h1*d4.y + h2v*d4.z + h3*d4.w; }
      }
      const float sc = 0.0625f;     // 1/sqrt(256)
      scoreL[r][q * 5 + 0] = s0 * sc;
      scoreL[r][q * 5 + 1] = s1 * sc;
      scoreL[r][q * 5 + 2] = s2 * sc;
      scoreL[r][q * 5 + 3] = s3 * sc;
      scoreL[r][q * 5 + 4] = s4 * sc;
      if(q == 0) doneL[r] = dacc;
    }
    __syncthreads();

    // ---- per-row decision (leader thread per row) ----
    if(tid < RWG){
      const int r = tid, b = bbase + r;
      float done_p = sigm(doneL[r] + db);
      int comp = compA[r]; int uniq = uniqA[r];
      unsigned vis = visA[r]; int first = firstA[r];
      int eligible = (!comp) && (uniq >= NNODE) && (t >= NNODE);
      int comp2 = comp || (eligible && (done_p > 0.5f));
      float s[NNODE];
      #pragma unroll
      for(int n = 0; n < NNODE; n++){
        float v = scoreL[r][n];
        if(!comp2 && (uniq < NNODE) && ((vis >> n) & 1u)) v = rp;
        if(comp2) v = (n == first) ? 100.0f : -1e9f;
        s[n] = v;
      }
      float m = s[0]; int am = 0;
      #pragma unroll
      for(int n = 1; n < NNODE; n++) if(s[n] > m){ m = s[n]; am = n; }
      float se = 0.f;
      #pragma unroll
      for(int n = 0; n < NNODE; n++) se += __expf(s[n] - m);
      int curr = comp2 ? first : am;          // curr is always the argmax
      float lp = __logf(1.f / se + 1e-10f);   // p = exp(s[curr]-m)/se = 1/se
      if(t == 0) first = curr;
      int newly = (!comp2) && !((vis >> curr) & 1u);
      uniq += newly;
      if(!comp2) vis |= (1u << curr);
      compA[r] = comp2; uniqA[r] = uniq; visA[r] = vis; firstA[r] = first; currA[r] = curr;
      tours[b * TOURW + t] = (float)curr;
      logps[b * TOURW + t] = lp;
    }
    __syncthreads();

    // ---- gather next LSTM input into Xx (float4 loads, packed bf16 stores) ----
    {
      const int c4 = (tid & 63) * 4;          // 64 column-groups of 4
      #pragma unroll 1
      for(int r = tid >> 6; r < RWG; r += 4){
        int curr = currA[r];
        float4 v = *(const float4*)(node_emb + ((bbase + r) * NNODE + curr) * DIM + c4);
        uint2 p;
        p.x = (unsigned)f2bf(v.x) | ((unsigned)f2bf(v.y) << 16);
        p.y = (unsigned)f2bf(v.z) | ((unsigned)f2bf(v.w) << 16);
        *(uint2*)(Xx + r * DIM + c4) = p;
      }
    }
    cur ^= 1;
  }
  __syncthreads();
  if(tid < RWG){
    int b = bbase + tid;
    tours[b * TOURW + MSTEPS] = compA[tid] ? 0.f : (float)firstA[tid];
    logps[b * TOURW + MSTEPS] = 0.f;
  }
}

// ---------------------------------------------------------------------------
extern "C" void kernel_launch(void* const* d_in, const int* in_sizes, int n_in,
                              void* d_out, int out_size, void* d_ws, size_t ws_size,
                              hipStream_t stream){
  const float* node_emb = (const float*)d_in[0];
  // d_in[1] = mask (unused by forward)
  const float* W_ih    = (const float*)d_in[2];
  const float* W_hh    = (const float*)d_in[3];
  const float* b_ih    = (const float*)d_in[4];
  const float* b_hh    = (const float*)d_in[5];
  const float* attn_w  = (const float*)d_in[6];
  const float* attn_b  = (const float*)d_in[7];
  const float* done_w  = (const float*)d_in[8];
  const float* done_b  = (const float*)d_in[9];
  const float* revisit = (const float*)d_in[10];
  // d_in[11] = greedy (always argmax path)

  char* ws = (char*)d_ws;
  unsigned short* wpack = (unsigned short*)ws;                          // 1 MB
  unsigned short* apack = (unsigned short*)(ws + (1 << 20));            // 128 KB
  float*          keys  = (float*)(ws + (1 << 20) + (1 << 17));         // 80 MB (B*N*D f32)
  float* out = (float*)d_out;

  pack_weights_kernel<<<2304, 256, 0, stream>>>(W_ih, W_hh, attn_w, wpack, apack);
  keys_kernel<<<640, 256, 0, stream>>>(node_emb, apack, attn_b, keys);
  decoder_kernel<<<BATCH / RWG, 256, 0, stream>>>(node_emb, wpack, b_ih, b_hh,
                                                  done_w, done_b, revisit, keys, out);
}
